// SelectiveSSM_56109452755117
// MI455X (gfx1250) — matrix operand order repaired and checked by reading, not compiled, not run on hardware
//
#include <hip/hip_runtime.h>

// ---------------- problem constants ----------------
#define D_MODEL 1024
#define D_STATE 16
#define D_CONV  4
#define D_INNER 2048          // D_MODEL * 2
#define BATCH   2
#define SEQ     2048
#define MTOK    (BATCH * SEQ) // 4096 tokens

// ---------------- WMMA / vector types -----------------------
typedef __attribute__((ext_vector_type(16))) __bf16 v16bf;
typedef __attribute__((ext_vector_type(8)))  float  v8f;

struct alignas(16) U4 { unsigned int x, y, z, w; };
struct U8 { U4 lo, hi; };            // 32 bytes == one bf16 WMMA fragment
struct alignas(16) F4 { float x, y, z, w; };

// pointer types matching the async Global->LDS builtin signature:
// (v4i32 addrspace(1)*, v4i32 addrspace(3)*, imm offset, imm cpol)
typedef int vi4 __attribute__((__vector_size__(16)));
typedef __attribute__((address_space(1))) vi4* g_vi4p;
typedef __attribute__((address_space(3))) vi4* l_vi4p;

#if defined(__has_builtin)
#if __has_builtin(__builtin_amdgcn_global_load_async_to_lds_b128) && \
    __has_builtin(__builtin_amdgcn_s_wait_asynccnt)
#define USE_ASYNC_LDS 1
#endif
#endif
#ifndef USE_ASYNC_LDS
#define USE_ASYNC_LDS 0
#endif

// one 16-byte Global -> LDS transfer (async DMA if available)
__device__ __forceinline__ void cp16_g2l(const unsigned short* g, U4* l) {
#if USE_ASYNC_LDS
    __builtin_amdgcn_global_load_async_to_lds_b128((g_vi4p)g, (l_vi4p)l, 0, 0);
#else
    *l = *(const U4*)g;
#endif
}

__device__ __forceinline__ void wait_async_le4() {
#if USE_ASYNC_LDS
    __builtin_amdgcn_s_wait_asynccnt((unsigned short)4);
#endif
}
__device__ __forceinline__ void wait_async_le0() {
#if USE_ASYNC_LDS
    __builtin_amdgcn_s_wait_asynccnt((unsigned short)0);
#endif
}

__device__ __forceinline__ unsigned short f32_to_bf16(float f) {
    unsigned int u = __float_as_uint(f);
    u += 0x7FFFu + ((u >> 16) & 1u);   // round-to-nearest-even
    return (unsigned short)(u >> 16);
}

__device__ __forceinline__ float silu_f(float x) {
    return x / (1.0f + __expf(-x));
}

// ---------------- f32 -> bf16 convert (vector x4) ----------------
__global__ __launch_bounds__(256) void k_cvt_bf16(const float* __restrict__ src,
                                                  unsigned short* __restrict__ dst,
                                                  int n4) {
    int i = blockIdx.x * 256 + threadIdx.x;
    if (i >= n4) return;
    F4 v = ((const F4*)src)[i];
    unsigned long long p =
        (unsigned long long)f32_to_bf16(v.x)
      | ((unsigned long long)f32_to_bf16(v.y) << 16)
      | ((unsigned long long)f32_to_bf16(v.z) << 32)
      | ((unsigned long long)f32_to_bf16(v.w) << 48);
    *(unsigned long long*)(dst + 4 * (size_t)i) = p;
}

// ---------------- transpose f32[K][N] -> bf16[N][K] via LDS ----------------
__global__ __launch_bounds__(256) void k_transpose_bf16(const float* __restrict__ src,
                                                        unsigned short* __restrict__ dst,
                                                        int K, int N) {
    __shared__ float tile[32][33];    // pad to kill bank conflicts
    int tx = threadIdx.x & 31;
    int ty = threadIdx.x >> 5;        // 0..7
    int n0 = blockIdx.x * 32;
    int k0 = blockIdx.y * 32;
#pragma unroll
    for (int i = 0; i < 4; ++i) {
        int k = k0 + ty + i * 8;
        tile[ty + i * 8][tx] = src[(size_t)k * N + n0 + tx];
    }
    __syncthreads();
#pragma unroll
    for (int i = 0; i < 4; ++i) {
        int n = n0 + ty + i * 8;
        dst[(size_t)n * K + k0 + tx] = f32_to_bf16(tile[tx][ty + i * 8]);
    }
}

// ---------------- bf16 WMMA GEMM: C = A[M][K] * Bt[N][K]^T ----------------
// 128x128 block tile, 8 waves (2x4 grid), each wave 64x32 = 4x2 WMMA 16x16x32.
// Operand tiles double-buffered in LDS, fed by async Global->LDS DMA
// (ASYNCcnt pipelined one tile ahead).  Row pitch 80B -> conflict-free
// ds_load_b128 fragment reads.
#define PADU4 5   // LDS row pitch in 16B units (5*16 = 80 bytes)

__global__ __launch_bounds__(256) void k_gemm_bf16(const unsigned short* __restrict__ A,
                                                   const unsigned short* __restrict__ Bt,
                                                   float* __restrict__ C0,
                                                   float* __restrict__ C1,
                                                   int M, int N, int K,
                                                   int split, int ldc0, int ldc1) {
    __shared__ U4 ldsA[2][128 * PADU4];   // 2 x 10 KB
    __shared__ U4 ldsB[2][128 * PADU4];   // 2 x 10 KB

    const int tid  = threadIdx.x;
    const int lane = tid & 31;
    const int wave = tid >> 5;
    const int wr   = wave >> 2;           // 0..1
    const int wc   = wave & 3;            // 0..3
    const int rowBlk = blockIdx.y * 128;
    const int colBlk = blockIdx.x * 128;
    const int lhalf = lane >> 4;          // 0 | 1
    const int lrow  = lane & 15;

    // loader mapping: 256 threads x 16B = 4KB/instruction-round; 2 rounds per 8KB tile
    const int ldRow   = tid >> 2;         // 0..63
    const int ldChunk = tid & 3;          // 0..3 (16B chunks of a 64B row-slab)
    const unsigned short* gA0 = A  + (size_t)(rowBlk + ldRow)      * K + ldChunk * 8;
    const unsigned short* gA1 = A  + (size_t)(rowBlk + ldRow + 64) * K + ldChunk * 8;
    const unsigned short* gB0 = Bt + (size_t)(colBlk + ldRow)      * K + ldChunk * 8;
    const unsigned short* gB1 = Bt + (size_t)(colBlk + ldRow + 64) * K + ldChunk * 8;

    auto issue_tile = [&](int k0, int buf) {
        cp16_g2l(gA0 + k0, &ldsA[buf][ldRow * PADU4 + ldChunk]);
        cp16_g2l(gA1 + k0, &ldsA[buf][(ldRow + 64) * PADU4 + ldChunk]);
        cp16_g2l(gB0 + k0, &ldsB[buf][ldRow * PADU4 + ldChunk]);
        cp16_g2l(gB1 + k0, &ldsB[buf][(ldRow + 64) * PADU4 + ldChunk]);
        // branch-free clamped prefetch of the K-slab after next
        int kpf = k0 + 32; kpf = (kpf > K - 32) ? (K - 32) : kpf;
        __builtin_prefetch(gA0 + kpf, 0, 3);
        __builtin_prefetch(gB0 + kpf, 0, 3);
    };

    v8f acc[4][2];
#pragma unroll
    for (int i = 0; i < 4; ++i)
#pragma unroll
        for (int j = 0; j < 2; ++j)
            acc[i][j] = (v8f){0.f, 0.f, 0.f, 0.f, 0.f, 0.f, 0.f, 0.f};

    issue_tile(0, 0);                      // prologue: tile 0 in flight

    const int nk = K >> 5;
    for (int kb = 0; kb < nk; ++kb) {
        const int cur = kb & 1;
        if (kb + 1 < nk) {                 // keep next tile in flight
            issue_tile((kb + 1) << 5, cur ^ 1);
            wait_async_le4();              // current tile's 4 transfers landed
        } else {
            wait_async_le0();
        }
        __syncthreads();                   // all waves' transfers visible

        // A fragments: 16x32 bf16; lanes 0-15 row m K{0..7,16..23}, lanes 16-31 K{8..15,24..31}
        v16bf afrag[4];
#pragma unroll
        for (int i = 0; i < 4; ++i) {
            const U4* p = &ldsA[cur][(wr * 64 + i * 16 + lrow) * PADU4];
            U8 f; f.lo = p[lhalf]; f.hi = p[2 + lhalf];
            afrag[i] = __builtin_bit_cast(v16bf, f);
        }
        // B fragments: lane n holds one column; lanes 0-15 K{0..7,16..23}, lanes 16-31 K{8..15,24..31}
        v16bf bfrag[2];
#pragma unroll
        for (int j = 0; j < 2; ++j) {
            const U4* p = &ldsB[cur][(wc * 32 + j * 16 + lrow) * PADU4];
            U8 f; f.lo = p[lhalf]; f.hi = p[2 + lhalf];
            bfrag[j] = __builtin_bit_cast(v16bf, f);
        }
#pragma unroll
        for (int i = 0; i < 4; ++i)
#pragma unroll
            for (int j = 0; j < 2; ++j)
                acc[i][j] = __builtin_amdgcn_wmma_f32_16x16x32_bf16(
                    false, afrag[i], false, bfrag[j], (short)0, acc[i][j],
                    false, false);
        __syncthreads();                   // reads done before buffer reuse
    }

    // ---- store: VGPR r <-> M = r + 8*lhalf, N = lrow
#pragma unroll
    for (int i = 0; i < 4; ++i) {
#pragma unroll
        for (int j = 0; j < 2; ++j) {
            int m0 = rowBlk + wr * 64 + i * 16 + 8 * lhalf;
            int n  = colBlk + wc * 32 + j * 16 + lrow;
            if (n < split) {
                float* cp = C0 + (size_t)m0 * ldc0 + n;
#pragma unroll
                for (int r = 0; r < 8; ++r) cp[(size_t)r * ldc0] = acc[i][j][r];
            } else {
                float* cp = C1 + (size_t)m0 * ldc1 + (n - split);
#pragma unroll
                for (int r = 0; r < 8; ++r) cp[(size_t)r * ldc1] = acc[i][j][r];
            }
        }
    }
}

// ---------------- causal depthwise conv (width 4) + bias + SiLU ----------------
__global__ __launch_bounds__(256) void k_conv_silu(const float* __restrict__ xc,
                                                   const float* __restrict__ w,
                                                   const float* __restrict__ bias,
                                                   float* __restrict__ out) {
    int idx = blockIdx.x * 256 + threadIdx.x;   // over 4096*2048
    int d = idx & (D_INNER - 1);
    int t = idx >> 11;                          // global token
    int tb = t & (SEQ - 1);                     // position inside batch (causal pad per batch)
    float acc = bias[d];
#pragma unroll
    for (int j = 0; j < D_CONV; ++j) {
        int tt = tb - (D_CONV - 1) + j;
        float v = (tt >= 0) ? xc[(size_t)(t - (D_CONV - 1) + j) * D_INNER + d] : 0.f;
        acc += w[d * D_CONV + j] * v;
    }
    out[idx] = silu_f(acc);
}

// ---------------- x_dbl = xc_act @ W_x  (N = 33, LDS-staged row) ----------------
__global__ __launch_bounds__(256) void k_xdbl(const float* __restrict__ xa,
                                              const float* __restrict__ Wx,
                                              float* __restrict__ xdbl) {
    __shared__ float row[D_INNER];
    int t = blockIdx.x;
    for (int d = threadIdx.x; d < D_INNER; d += 256)
        row[d] = xa[(size_t)t * D_INNER + d];
    __syncthreads();
    int c = threadIdx.x;
    if (c < 2 * D_STATE + 1) {
        float acc = 0.f;
        for (int d = 0; d < D_INNER; ++d)
            acc += row[d] * Wx[d * (2 * D_STATE + 1) + c];
        xdbl[(size_t)t * (2 * D_STATE + 1) + c] = acc;
    }
}

// ---------------- selective scan + residual + gate, emit bf16 for final GEMM ----
__global__ __launch_bounds__(256) void k_scan(const float* __restrict__ xa,
                                              const float* __restrict__ zb,
                                              const float* __restrict__ xdbl,
                                              const float* __restrict__ Wdt,
                                              const float* __restrict__ bdt,
                                              const float* __restrict__ Alog,
                                              const float* __restrict__ Dp,
                                              unsigned short* __restrict__ ybf) {
    int idx = blockIdx.x * 256 + threadIdx.x;   // 0..4095 : (batch, channel)
    int d = idx & (D_INNER - 1);
    int b = idx >> 11;

    float A[D_STATE];
#pragma unroll
    for (int s = 0; s < D_STATE; ++s) A[s] = -__expf(Alog[d * D_STATE + s]);
    const float wdt = Wdt[d], bdtv = bdt[d], Dv = Dp[d];

    float h[D_STATE];
#pragma unroll
    for (int s = 0; s < D_STATE; ++s) h[s] = 0.f;

    for (int t = 0; t < SEQ; ++t) {
        size_t row = (size_t)(b * SEQ + t);
        const float* xd = xdbl + row * (2 * D_STATE + 1);
        float pre = xd[0] * wdt + bdtv;
        float dt = (pre > 20.f) ? pre : __logf(1.f + __expf(pre));  // softplus
        float xt = xa[row * D_INNER + d];
        float y = 0.f;
#pragma unroll
        for (int s = 0; s < D_STATE; ++s) {
            float dA = __expf(dt * A[s]);
            h[s] = dA * h[s] + (dt * xt) * xd[1 + s];
            y += h[s] * xd[1 + D_STATE + s];
        }
        float yo = y + Dv * xt;
        float zv = zb[row * D_INNER + d];
        ybf[row * D_INNER + d] = f32_to_bf16(yo * silu_f(zv));
    }
}

// ---------------- host-side orchestration ----------------
extern "C" void kernel_launch(void* const* d_in, const int* in_sizes, int n_in,
                              void* d_out, int out_size, void* d_ws, size_t ws_size,
                              hipStream_t stream) {
    const float* x      = (const float*)d_in[0];
    const float* W_in   = (const float*)d_in[1];
    const float* conv_w = (const float*)d_in[2];
    const float* conv_b = (const float*)d_in[3];
    const float* W_x    = (const float*)d_in[4];
    const float* W_dt   = (const float*)d_in[5];
    const float* b_dt   = (const float*)d_in[6];
    const float* A_log  = (const float*)d_in[7];
    const float* D_par  = (const float*)d_in[8];
    const float* W_out  = (const float*)d_in[9];
    float* out = (float*)d_out;

    // workspace carve-up (~133 MB)
    unsigned short* xb    = (unsigned short*)d_ws;                   // x as bf16     [4096][1024]
    unsigned short* WtIn  = xb    + (size_t)MTOK * D_MODEL;          // W_in^T bf16   [4096][1024]
    unsigned short* WtOut = WtIn  + (size_t)(2 * D_INNER) * D_MODEL; // W_out^T bf16  [1024][2048]
    unsigned short* ybf   = WtOut + (size_t)D_MODEL * D_INNER;       // gated y bf16  [4096][2048]
    float* xc   = (float*)(ybf + (size_t)MTOK * D_INNER);            // pre-conv xc   [4096][2048]
    float* zb   = xc   + (size_t)MTOK * D_INNER;                     // z             [4096][2048]
    float* xa   = zb   + (size_t)MTOK * D_INNER;                     // conv+SiLU xc  [4096][2048]
    float* xdbl = xa   + (size_t)MTOK * D_INNER;                     // x_dbl         [4096][33]

    // 1) bf16 conversions / weight transposes
    k_cvt_bf16<<<(MTOK * D_MODEL / 4 + 255) / 256, 256, 0, stream>>>(x, xb, MTOK * D_MODEL / 4);
    k_transpose_bf16<<<dim3((2 * D_INNER) / 32, D_MODEL / 32), 256, 0, stream>>>(W_in, WtIn, D_MODEL, 2 * D_INNER);
    k_transpose_bf16<<<dim3(D_MODEL / 32, D_INNER / 32), 256, 0, stream>>>(W_out, WtOut, D_INNER, D_MODEL);

    // 2) xz = x @ W_in  (M=4096, N=4096, K=1024) -> xc | z split at N=2048
    k_gemm_bf16<<<dim3((2 * D_INNER) / 128, MTOK / 128), 256, 0, stream>>>(
        xb, WtIn, xc, zb, MTOK, 2 * D_INNER, D_MODEL, D_INNER, D_INNER, D_INNER);

    // 3) causal depthwise conv + SiLU
    k_conv_silu<<<(MTOK * D_INNER) / 256, 256, 0, stream>>>(xc, conv_w, conv_b, xa);

    // 4) x_dbl = xc_act @ W_x  (N = 33)
    k_xdbl<<<MTOK, 256, 0, stream>>>(xa, W_x, xdbl);

    // 5) selective scan + D-residual + SiLU(z) gate -> bf16
    k_scan<<<MTOK / 256, 256, 0, stream>>>(xa, zb, xdbl, W_dt, b_dt, A_log, D_par, ybf);

    // 6) out = y @ W_out  (M=4096, N=1024, K=2048)
    k_gemm_bf16<<<dim3(D_MODEL / 128, MTOK / 128), 256, 0, stream>>>(
        ybf, WtOut, out, out, MTOK, D_MODEL, D_INNER, D_MODEL, D_MODEL, D_MODEL);
}
